// ED_Pointnet_Backbone_89601607729775
// MI455X (gfx1250) — matrix operand order repaired and checked by reading, not compiled, not run on hardware
//
#include <hip/hip_runtime.h>
#include <hip/hip_bf16.h>
#include <math.h>

// ---------------------------------------------------------------------------
// ED_Pointnet_Backbone for MI455X (gfx1250, wave32, WMMA)
//
//  * All intermediates L2-resident (192MB); HBM BW irrelevant.
//  * GEMMs: v_wmma_f32_16x16x32_f16. Activations kept in f16 with K-padded
//    row stride -> A fragments are 2x b128 contiguous loads per lane.
//    Weights pre-repacked once to f16 [N][KP] -> B fragments 2x b128 loads.
//    KP templated -> fully unrolled wmma chains. f32 accumulate + f32 global
//    BatchNorm statistics (atomics) preserve accuracy.
//  * KNN fused distance+top-k (8x4096x4096 never materialized).
//  * FPS: 1 workgroup/batch, LDS-resident distance array.
// ---------------------------------------------------------------------------

typedef __attribute__((ext_vector_type(16))) _Float16 v16h;
typedef __attribute__((ext_vector_type(8)))  _Float16 v8h;
typedef __attribute__((ext_vector_type(8)))  float    v8f;

#define BATCH 8
#define NP0   4096
#define NP1   1024
#define NP2   512
#define NP3   256
#define NS1   32
#define NS2   48
#define NS3   48
#define EDK   10
#define EPS_BN 1e-5f

#define MMF_STATS 1
#define MMF_BIAS  2

// ------------------------------ utility kernels ----------------------------

__global__ void zero_kernel(float* __restrict__ p, int n) {
  for (int i = blockIdx.x * blockDim.x + threadIdx.x; i < n;
       i += blockDim.x * gridDim.x) p[i] = 0.0f;
}

__global__ void copy_kernel(float* __restrict__ dst, const float* __restrict__ src,
                            long long n) {
  long long i = (long long)blockIdx.x * blockDim.x + threadIdx.x;
  if (i < n) dst[i] = src[i];
}

__global__ void cvt_f16_kernel(const float* __restrict__ src, _Float16* __restrict__ dst,
                               long long n) {
  long long i = (long long)blockIdx.x * blockDim.x + threadIdx.x;
  if (i < n) dst[i] = (_Float16)src[i];
}

// Repack weights W[K][N] f32 -> Wp[N][KP] f16 (zero pad K..KP).
__global__ void repack_w_kernel(const float* __restrict__ W, int K, int N, int KP,
                                _Float16* __restrict__ Wp) {
  int i = blockIdx.x * blockDim.x + threadIdx.x;  // N*KP
  if (i >= N * KP) return;
  int k = i % KP, n = i / KP;
  Wp[i] = (k < K) ? (_Float16)W[(size_t)k * N + n] : (_Float16)0.0f;
}

// ------------------------------ FPS ----------------------------------------
__global__ __launch_bounds__(256)
void fps_kernel(const float* __restrict__ xyz, int N, int npoint,
                int* __restrict__ fidx) {
  __shared__ float dist[4096];
  __shared__ float rmx[256];
  __shared__ int   rix[256];
  __shared__ int   sfar;
  const int b = blockIdx.x;
  const int t = threadIdx.x;
  const float* px = xyz + (size_t)b * N * 3;
  for (int i = t; i < N; i += 256) dist[i] = 1e10f;
  if (t == 0) sfar = 0;
  __syncthreads();
  for (int j = 0; j < npoint; ++j) {
    int far = sfar;
    if (t == 0) fidx[b * npoint + j] = far;
    float cx = px[far * 3 + 0], cy = px[far * 3 + 1], cz = px[far * 3 + 2];
    float lm = -1.0f; int li = 0;
    for (int i = t; i < N; i += 256) {
      float dx = px[i * 3 + 0] - cx, dy = px[i * 3 + 1] - cy, dz = px[i * 3 + 2] - cz;
      float d = dx * dx + dy * dy + dz * dz;
      float o = dist[i];
      d = d < o ? d : o;
      dist[i] = d;
      if (d > lm) { lm = d; li = i; }   // strict > keeps earliest index
    }
    rmx[t] = lm; rix[t] = li;
    __syncthreads();
    for (int s = 128; s > 0; s >>= 1) {
      if (t < s) {
        if (rmx[t + s] > rmx[t] ||
            (rmx[t + s] == rmx[t] && rix[t + s] < rix[t])) {
          rmx[t] = rmx[t + s]; rix[t] = rix[t + s];
        }
      }
      __syncthreads();
    }
    if (t == 0) sfar = rix[0];
    __syncthreads();
  }
}

__global__ void gather3_kernel(const float* __restrict__ xyz, const int* __restrict__ fidx,
                               int N, int S, float* __restrict__ out) {
  long long i = (long long)blockIdx.x * blockDim.x + threadIdx.x;  // B*S
  if (i >= (long long)BATCH * S) return;
  int b = (int)(i / S);
  int j = fidx[i];
  const float* p = xyz + ((size_t)b * N + j) * 3;
  float* o = out + i * 3;
  o[0] = p[0]; o[1] = p[1]; o[2] = p[2];
}

// ------------------------------ KNN ----------------------------------------
__global__ __launch_bounds__(64)
void knn_kernel(const float* __restrict__ src, int N,
                const float* __restrict__ q, int S, int K,
                int* __restrict__ out) {
  __shared__ float bd[64 * 48];
  __shared__ int   bi[64 * 48];
  const int b = blockIdx.y;
  const int t = threadIdx.x;
  const int s = blockIdx.x * 64 + t;
  const float* ps = src + (size_t)b * N * 3;
  float qx = 0.f, qy = 0.f, qz = 0.f;
  if (s < S) {
    const float* pq = q + ((size_t)b * S + s) * 3;
    qx = pq[0]; qy = pq[1]; qz = pq[2];
  }
  for (int p = 0; p < K; ++p) { bd[p * 64 + t] = 3.0e38f; bi[p * 64 + t] = 0; }
  for (int j = 0; j < N; ++j) {
    __builtin_prefetch(ps + (j + 16) * 3, 0, 0);   // global_prefetch_b8
    float dx = ps[j * 3 + 0] - qx, dy = ps[j * 3 + 1] - qy, dz = ps[j * 3 + 2] - qz;
    float d = dx * dx + dy * dy + dz * dz;
    if (d < bd[(K - 1) * 64 + t]) {
      int p = K - 1;
      while (p > 0 && bd[(p - 1) * 64 + t] > d) {
        bd[p * 64 + t] = bd[(p - 1) * 64 + t];
        bi[p * 64 + t] = bi[(p - 1) * 64 + t];
        --p;
      }
      bd[p * 64 + t] = d; bi[p * 64 + t] = j;
    }
  }
  if (s < S) {
    int* po = out + ((size_t)b * S + s) * K;
    for (int p = 0; p < K; ++p) po[p] = bi[p * 64 + t];
  }
}

// --------------------------- grouping kernels (f16 out) --------------------

// SA1 input: rel xyz (3ch) zero-padded to stride 32.
__global__ void group_rel_kernel(const float* __restrict__ xyz,
                                 const float* __restrict__ nxyz,
                                 const int* __restrict__ nidx,
                                 int S, int Kn, int N, _Float16* __restrict__ X) {
  long long i = (long long)blockIdx.x * blockDim.x + threadIdx.x;  // B*S*Kn
  long long total = (long long)BATCH * S * Kn;
  if (i >= total) return;
  long long bs = i / Kn;
  int s = (int)(bs % S);
  int b = (int)(bs / S);
  int j = nidx[i];
  const float* p = xyz + ((size_t)b * N + j) * 3;
  const float* c = nxyz + ((size_t)b * S + s) * 3;
  _Float16* o = X + i * 32;
  o[0] = (_Float16)(p[0] - c[0]);
  o[1] = (_Float16)(p[1] - c[1]);
  o[2] = (_Float16)(p[2] - c[2]);
  for (int cc = 3; cc < 32; ++cc) o[cc] = (_Float16)0.0f;
}

// SA2/3 input: [center_feat, nb_feat - center_feat], stride 2C (mult of 32).
__global__ void group_feat_kernel(const _Float16* __restrict__ feats,
                                  const int* __restrict__ nidx,
                                  const int* __restrict__ fidx,
                                  int S, int Kn, int N, int C,
                                  _Float16* __restrict__ X) {
  long long i = (long long)blockIdx.x * blockDim.x + threadIdx.x;  // B*S*Kn
  long long total = (long long)BATCH * S * Kn;
  if (i >= total) return;
  long long bs = i / Kn;
  int s = (int)(bs % S);
  int b = (int)(bs / S);
  int j  = nidx[i];
  int jc = fidx[(size_t)b * S + s];
  const _Float16* fj = feats + ((size_t)b * N + j)  * C;
  const _Float16* fc = feats + ((size_t)b * N + jc) * C;
  _Float16* o = X + i * (size_t)(2 * C);
  for (int c = 0; c < C; ++c) {
    float cf = (float)fc[c];
    o[c]     = (_Float16)cf;
    o[C + c] = (_Float16)((float)fj[c] - cf);
  }
}

// --------------------------- WMMA GEMM (+BN stats) -------------------------
// Y[M,N] = X[M,KP] @ Wp[N,KP]^T (+bias). One wave per 16x16 tile. X,Wp f16
// with KP (mult of 32) stride; contiguous b128 fragment loads; fully
// unrolled wmma chain; f32 accumulate; optional per-column sum/sumsq atomics.
template <int KP>
__global__ __launch_bounds__(32)
void mm_wmma_kernel(const _Float16* __restrict__ X, const _Float16* __restrict__ Wp,
                    const float* __restrict__ bias, float* __restrict__ Y,
                    float* __restrict__ stats, int N, int flags) {
  const int lane = threadIdx.x;       // block == one wave32
  const int half = lane >> 4;
  const int l15  = lane & 15;
  const long long row0 = (long long)blockIdx.x * 16;
  const int col = blockIdx.y * 16 + l15;

  // A f16 16x32 layout: lane(m=l15): elems 0..7 -> k=kb+half*8+(0..7),
  //                                  elems 8..15 -> k=kb+16+half*8+(0..7)
  const _Float16* xa = X + (size_t)(row0 + l15) * KP + half * 8;
  // B f16 32x16 layout: lane(n=l15): elems 0..7 -> k=kb+half*8+(0..7),
  //                                  elems 8..15 -> k=kb+16+half*8+(0..7)
  const _Float16* wb = Wp + (size_t)col * KP + half * 8;

  v8f c = {};
#pragma unroll
  for (int kb = 0; kb < KP; kb += 32) {
    v8h a0 = *(const v8h*)(xa + kb);
    v8h a1 = *(const v8h*)(xa + kb + 16);
    v8h b0 = *(const v8h*)(wb + kb);
    v8h b1 = *(const v8h*)(wb + kb + 16);
    v16h af = __builtin_shufflevector(a0, a1, 0,1,2,3,4,5,6,7,8,9,10,11,12,13,14,15);
    v16h bf = __builtin_shufflevector(b0, b1, 0,1,2,3,4,5,6,7,8,9,10,11,12,13,14,15);
    c = __builtin_amdgcn_wmma_f32_16x16x32_f16(
        /*neg_a=*/false, af, /*neg_b=*/false, bf,
        /*c_mod=*/(short)0, c, /*reuse_a=*/false, /*reuse_b=*/false);
  }

  const float bv = (flags & MMF_BIAS) ? bias[col] : 0.0f;
  float s = 0.f, s2 = 0.f;
#pragma unroll
  for (int v = 0; v < 8; ++v) {
    long long r = row0 + (half << 3) + v;   // C/D: m = v + half*8, n = l15
    float y = c[v] + bv;
    Y[(size_t)r * N + col] = y;
    s += y; s2 += y * y;
  }
  if (flags & MMF_STATS) {
    atomicAdd(&stats[col], s);
    atomicAdd(&stats[N + col], s2);
  }
}

__global__ void bn_finalize_kernel(float* __restrict__ stats,
                                   const float* __restrict__ g,
                                   const float* __restrict__ b,
                                   int N, float invM) {
  int n = threadIdx.x;
  if (n >= N) return;
  float mu  = stats[n] * invM;
  float var = stats[N + n] * invM - mu * mu;
  var = var < 0.f ? 0.f : var;
  float sc = g[n] * rsqrtf(var + EPS_BN);
  stats[2 * N + n] = sc;
  stats[3 * N + n] = b[n] - mu * sc;
}

// scale+shift+ReLU, f32 pre-act -> f16 next-layer activations
__global__ void bn_relu_f16_kernel(const float* __restrict__ Y,
                                   const float* __restrict__ stats,
                                   _Float16* __restrict__ Xn,
                                   long long total, int N) {
  long long i = (long long)blockIdx.x * blockDim.x + threadIdx.x;
  if (i >= total) return;
  int n = (int)(i % N);
  float y = Y[i] * stats[2 * N + n] + stats[3 * N + n];
  Xn[i] = (_Float16)(y > 0.f ? y : 0.f);
}

// --------------------------- pooling / FP / head ---------------------------

__global__ void maxpool_kernel(const _Float16* __restrict__ Y, long long rows, int Kn,
                               int C, _Float16* __restrict__ out) {
  long long i = (long long)blockIdx.x * blockDim.x + threadIdx.x;  // rows*C
  if (i >= rows * C) return;
  int c = (int)(i % C);
  long long r = i / C;
  const _Float16* p = Y + r * (size_t)Kn * C + c;
  float m = (float)p[0];
  for (int kk = 1; kk < Kn; ++kk) {
    float v = (float)p[(size_t)kk * C];
    m = v > m ? v : m;
  }
  out[i] = (_Float16)m;
}

// 3-NN inverse-distance interpolation + concat [f1, interp], f16 out, pad->KP
__global__ void fp_cat_kernel(const float* __restrict__ qxyz, const float* __restrict__ sxyz,
                              const int* __restrict__ idx,
                              const _Float16* __restrict__ f1, int C1,
                              const _Float16* __restrict__ f2, int C2,
                              int Sq, int Ns, int KP, _Float16* __restrict__ cat) {
  long long i = (long long)blockIdx.x * blockDim.x + threadIdx.x;  // B*Sq
  if (i >= (long long)BATCH * Sq) return;
  int b = (int)(i / Sq);
  const float* pq = qxyz + i * 3;
  const int* pid = idx + i * 3;
  float w[3]; float ws = 0.f;
  for (int jj = 0; jj < 3; ++jj) {
    const float* pp = sxyz + ((size_t)b * Ns + pid[jj]) * 3;
    float dx = pp[0] - pq[0], dy = pp[1] - pq[1], dz = pp[2] - pq[2];
    float d = dx * dx + dy * dy + dz * dz;
    w[jj] = 1.0f / (d + 1e-8f);
    ws += w[jj];
  }
  float inv = 1.0f / ws;
  _Float16* o = cat + i * (size_t)KP;
  const _Float16* p1 = f1 + i * (size_t)C1;
  for (int c = 0; c < C1; ++c) o[c] = p1[c];
  for (int c = 0; c < C2; ++c) {
    float acc = 0.f;
    for (int jj = 0; jj < 3; ++jj)
      acc += w[jj] * (float)f2[((size_t)b * Ns + pid[jj]) * C2 + c];
    o[C1 + c] = (_Float16)(acc * inv);
  }
  for (int c = C1 + C2; c < KP; ++c) o[c] = (_Float16)0.0f;
}

// cov of 10-NN -> analytic eigvals of symmetric 3x3 (ascending) -> 3->8->8 MLP
__global__ void eig_kernel(const float* __restrict__ xyz, const int* __restrict__ gidx,
                           const float* __restrict__ W1, const float* __restrict__ b1,
                           const float* __restrict__ W2, const float* __restrict__ b2,
                           float* __restrict__ ef) {
  long long i = (long long)blockIdx.x * blockDim.x + threadIdx.x;  // B*NP0
  if (i >= (long long)BATCH * NP0) return;
  int bb = (int)(i / NP0);
  const int* pid = gidx + i * EDK;
  const float* px = xyz + (size_t)bb * NP0 * 3;
  float pts[EDK][3];
  float mx = 0.f, my = 0.f, mz = 0.f;
  for (int k = 0; k < EDK; ++k) {
    const float* p = px + pid[k] * 3;
    pts[k][0] = p[0]; pts[k][1] = p[1]; pts[k][2] = p[2];
    mx += p[0]; my += p[1]; mz += p[2];
  }
  mx *= 0.1f; my *= 0.1f; mz *= 0.1f;
  float a00 = 0, a01 = 0, a02 = 0, a11 = 0, a12 = 0, a22 = 0;
  for (int k = 0; k < EDK; ++k) {
    float x = pts[k][0] - mx, y = pts[k][1] - my, z = pts[k][2] - mz;
    a00 += x * x; a01 += x * y; a02 += x * z;
    a11 += y * y; a12 += y * z; a22 += z * z;
  }
  a00 *= 0.1f; a01 *= 0.1f; a02 *= 0.1f; a11 *= 0.1f; a12 *= 0.1f; a22 *= 0.1f;
  float ev0, ev1, ev2;
  float p1 = a01 * a01 + a02 * a02 + a12 * a12;
  float qt = (a00 + a11 + a22) * (1.0f / 3.0f);
  float p2 = (a00 - qt) * (a00 - qt) + (a11 - qt) * (a11 - qt) +
             (a22 - qt) * (a22 - qt) + 2.0f * p1;
  if (p2 <= 1e-30f) {
    ev0 = ev1 = ev2 = qt;
  } else {
    float p = sqrtf(p2 * (1.0f / 6.0f));
    float ip = 1.0f / p;
    float c00 = (a00 - qt) * ip, c11 = (a11 - qt) * ip, c22 = (a22 - qt) * ip;
    float c01 = a01 * ip, c02 = a02 * ip, c12 = a12 * ip;
    float detB = c00 * (c11 * c22 - c12 * c12)
               - c01 * (c01 * c22 - c12 * c02)
               + c02 * (c01 * c12 - c11 * c02);
    float r = detB * 0.5f;
    r = r < -1.f ? -1.f : (r > 1.f ? 1.f : r);
    float phi = acosf(r) * (1.0f / 3.0f);
    float e1 = qt + 2.0f * p * cosf(phi);                       // largest
    float e3 = qt + 2.0f * p * cosf(phi + 2.0943951023931953f); // smallest
    float e2 = 3.0f * qt - e1 - e3;
    ev0 = e3; ev1 = e2; ev2 = e1;  // eigvalsh ascending
  }
  float h[8];
  for (int o = 0; o < 8; ++o) {
    float v = ev0 * W1[0 * 8 + o] + ev1 * W1[1 * 8 + o] + ev2 * W1[2 * 8 + o] + b1[o];
    h[o] = v > 0.f ? v : 0.f;
  }
  float* po = ef + i * 8;
  for (int o = 0; o < 8; ++o) {
    float v = b2[o];
    for (int u = 0; u < 8; ++u) v += h[u] * W2[u * 8 + o];
    po[o] = v;
  }
}

// concat [out2(32), ef(8)] -> f16, stride 64 (zero pad 40..63)
__global__ void concat_pad_kernel(const float* __restrict__ o32, const float* __restrict__ e8,
                                  _Float16* __restrict__ z) {
  long long i = (long long)blockIdx.x * blockDim.x + threadIdx.x;  // B*NP0
  if (i >= (long long)BATCH * NP0) return;
  _Float16* o = z + i * 64;
  const float* a = o32 + i * 32;
  const float* e = e8 + i * 8;
  for (int c = 0; c < 32; ++c) o[c] = (_Float16)a[c];
  for (int c = 0; c < 8; ++c) o[32 + c] = (_Float16)e[c];
  for (int c = 40; c < 64; ++c) o[c] = (_Float16)0.0f;
}

// fused BN + ReLU + transpose(0,2,1) into final output
__global__ void bn_relu_transpose_kernel(const float* __restrict__ z,
                                         const float* __restrict__ stats,
                                         float* __restrict__ out) {
  long long i = (long long)blockIdx.x * blockDim.x + threadIdx.x;  // B*NP0*32
  if (i >= (long long)BATCH * NP0 * 32) return;
  int c = (int)(i % 32);
  long long bn = i / 32;
  int n = (int)(bn % NP0);
  int b = (int)(bn / NP0);
  float y = z[i] * stats[64 + c] + stats[96 + c];
  y = y > 0.f ? y : 0.f;
  out[((size_t)b * 32 + c) * NP0 + n] = y;
}

// ------------------------------- host side ---------------------------------

namespace {

struct Bump {
  char* p; size_t off, cap; bool ok;
  void* take(size_t bytes) {
    size_t a = (off + 255) & ~(size_t)255;
    if (a + bytes > cap) { ok = false; return p; }
    off = a + bytes;
    return p + a;
  }
  float*    f(size_t n) { return (float*)take(n * sizeof(float)); }
  int*      i(size_t n) { return (int*)take(n * sizeof(int)); }
  _Float16* h(size_t n) { return (_Float16*)take(n * sizeof(_Float16)); }
};

inline unsigned gdiv(long long n, int b) { return (unsigned)((n + b - 1) / b); }

struct Layer { const float *W, *b, *g; };

} // namespace

extern "C" void kernel_launch(void* const* d_in, const int* in_sizes, int n_in,
                              void* d_out, int out_size, void* d_ws, size_t ws_size,
                              hipStream_t stream) {
  (void)out_size;
  const float* xyz0 = (const float*)d_in[0];

  // 'numpoints' may be one array of 3 ints or three scalars; skip it.
  int pi = (n_in >= 2 && in_sizes[1] == 3) ? 2 : 4;

  const float *bn1_b, *bn1_g, *conv1_W, *conv1_b, *cov_W, *cov_b;
  const float *ed1_W, *ed1_b, *ed2_W, *ed2_b;
  Layer fp[3][2], sa[3][3];

  if (pi < n_in && in_sizes[pi] == 32) {
    // jax pytree flatten: dict keys sorted.
    bn1_b = (const float*)d_in[pi++]; bn1_g = (const float*)d_in[pi++];
    conv1_W = (const float*)d_in[pi++]; conv1_b = (const float*)d_in[pi++];
    cov_W = (const float*)d_in[pi++]; cov_b = (const float*)d_in[pi++];
    ed1_W = (const float*)d_in[pi++]; ed1_b = (const float*)d_in[pi++];
    ed2_W = (const float*)d_in[pi++]; ed2_b = (const float*)d_in[pi++];
    for (int i = 0; i < 3; ++i)
      for (int l = 0; l < 2; ++l) {
        fp[i][l].W = (const float*)d_in[pi++];
        fp[i][l].b = (const float*)d_in[pi++];
        fp[i][l].g = (const float*)d_in[pi++];
      }
    for (int i = 0; i < 3; ++i)
      for (int l = 0; l < 3; ++l) {
        sa[i][l].W = (const float*)d_in[pi++];
        sa[i][l].b = (const float*)d_in[pi++];
        sa[i][l].g = (const float*)d_in[pi++];
      }
  } else {
    // insertion-order flatten fallback.
    for (int i = 0; i < 3; ++i)
      for (int l = 0; l < 3; ++l) {
        sa[i][l].W = (const float*)d_in[pi++];
        sa[i][l].g = (const float*)d_in[pi++];
        sa[i][l].b = (const float*)d_in[pi++];
      }
    for (int i = 0; i < 3; ++i)
      for (int l = 0; l < 2; ++l) {
        fp[i][l].W = (const float*)d_in[pi++];
        fp[i][l].g = (const float*)d_in[pi++];
        fp[i][l].b = (const float*)d_in[pi++];
      }
    cov_W = (const float*)d_in[pi++]; cov_b = (const float*)d_in[pi++];
    ed1_W = (const float*)d_in[pi++]; ed1_b = (const float*)d_in[pi++];
    ed2_W = (const float*)d_in[pi++]; ed2_b = (const float*)d_in[pi++];
    conv1_W = (const float*)d_in[pi++]; conv1_b = (const float*)d_in[pi++];
    bn1_g = (const float*)d_in[pi++]; bn1_b = (const float*)d_in[pi++];
  }

  // ---- workspace ----
  Bump A{(char*)d_ws, 0, ws_size, true};
  const size_t BIG = (size_t)BATCH * NP2 * NS2 * 64;  // 12.58M elems (max rows*C)
  float* xyz1 = A.f((size_t)BATCH * NP1 * 3);
  float* xyz2 = A.f((size_t)BATCH * NP2 * 3);
  float* xyz3 = A.f((size_t)BATCH * NP3 * 3);
  _Float16* xyz0h = A.h((size_t)BATCH * NP0 * 3);
  int* fidx1 = A.i((size_t)BATCH * NP1);
  int* fidx2 = A.i((size_t)BATCH * NP2);
  int* fidx3 = A.i((size_t)BATCH * NP3);
  int* nidx1 = A.i((size_t)BATCH * NP1 * NS1);
  int* nidx2 = A.i((size_t)BATCH * NP2 * NS2);
  int* nidx3 = A.i((size_t)BATCH * NP3 * NS3);
  int* fpidx2 = A.i((size_t)BATCH * NP2 * 3);
  int* fpidx1 = A.i((size_t)BATCH * NP1 * 3);
  int* fpidx0 = A.i((size_t)BATCH * NP0 * 3);
  int* gidx = A.i((size_t)BATCH * NP0 * EDK);
  float* Yf = A.f(BIG);          // f32 pre-activation GEMM output
  _Float16* hA = A.h(BIG);       // f16 activation ping
  _Float16* hB = A.h(BIG);       // f16 activation pong
  _Float16* feat1 = A.h((size_t)BATCH * NP1 * 32);
  _Float16* feat2 = A.h((size_t)BATCH * NP2 * 64);
  _Float16* feat3 = A.h((size_t)BATCH * NP3 * 128);
  _Float16* fpf2 = A.h((size_t)BATCH * NP2 * 128);
  _Float16* fpf1 = A.h((size_t)BATCH * NP1 * 64);
  _Float16* fpf0 = A.h((size_t)BATCH * NP0 * 32);
  float* efbuf = A.f((size_t)BATCH * NP0 * 8);
  float* stats = A.f(512);
  // head overlays (Yf / hA / hB are free by then)
  float* out2buf = Yf;                       // B*NP0*32 f32
  _Float16* zbuf = hA;                       // B*NP0*64 f16 (padded 40->64)
  float* z2buf = (float*)hB;                 // B*NP0*32 f32

  // packed f16 weights [N][KP]
  auto pack = [&](const Layer& L, int K, int N, int KP) -> _Float16* {
    _Float16* w = A.h((size_t)N * KP);
    if (A.ok)
      repack_w_kernel<<<gdiv((long long)N * KP, 256), 256, 0, stream>>>(L.W, K, N, KP, w);
    return w;
  };
  // KP per layer (K rounded up to 32)
  _Float16* saP[3][3];
  _Float16* fpP[3][2];
  const int saK[3][3] = {{3, 32, 32}, {64, 64, 64}, {128, 128, 128}};
  const int saN[3] = {32, 64, 128};
  const int fpK[3][2] = {{67, 32}, {160, 128}, {192, 128}};
  const int fpN[3][2] = {{32, 32}, {128, 64}, {128, 128}};
  for (int i = 0; i < 3; ++i)
    for (int l = 0; l < 3; ++l)
      saP[i][l] = pack(sa[i][l], saK[i][l], saN[i], (saK[i][l] + 31) & ~31);
  for (int i = 0; i < 3; ++i)
    for (int l = 0; l < 2; ++l)
      fpP[i][l] = pack(fp[i][l], fpK[i][l], fpN[i][l], (fpK[i][l] + 31) & ~31);
  Layer covL{cov_W, cov_b, nullptr}, conv1L{conv1_W, conv1_b, nullptr};
  _Float16* covP = pack(covL, 32, 32, 32);
  _Float16* conv1P = pack(conv1L, 40, 32, 64);
  if (!A.ok) return;

  auto run_linear = [&](const _Float16* X, int KP, const _Float16* Wp,
                        const float* bias, float* Y, long long M, int N, int flags) {
    dim3 g((unsigned)(M / 16), (unsigned)(N / 16));
    switch (KP) {
      case 32:  mm_wmma_kernel<32><<<g, 32, 0, stream>>>(X, Wp, bias, Y, stats, N, flags); break;
      case 64:  mm_wmma_kernel<64><<<g, 32, 0, stream>>>(X, Wp, bias, Y, stats, N, flags); break;
      case 96:  mm_wmma_kernel<96><<<g, 32, 0, stream>>>(X, Wp, bias, Y, stats, N, flags); break;
      case 128: mm_wmma_kernel<128><<<g, 32, 0, stream>>>(X, Wp, bias, Y, stats, N, flags); break;
      case 160: mm_wmma_kernel<160><<<g, 32, 0, stream>>>(X, Wp, bias, Y, stats, N, flags); break;
      case 192: mm_wmma_kernel<192><<<g, 32, 0, stream>>>(X, Wp, bias, Y, stats, N, flags); break;
      default: break;
    }
  };
  auto run_mlp_layer = [&](const _Float16* X, int KP, const _Float16* Wp,
                           const Layer& L, _Float16* Xout, long long M, int N) {
    zero_kernel<<<1, 256, 0, stream>>>(stats, 2 * N);
    run_linear(X, KP, Wp, nullptr, Yf, M, N, MMF_STATS);
    bn_finalize_kernel<<<1, 128, 0, stream>>>(stats, L.g, L.b, N, 1.0f / (float)M);
    long long tot = M * (long long)N;
    bn_relu_f16_kernel<<<gdiv(tot, 256), 256, 0, stream>>>(Yf, stats, Xout, tot, N);
  };

  float* out1 = (float*)d_out;
  float* outz = out1 + (size_t)BATCH * NP0 * 3;

  // out1 = xyz; f16 copy of xyz for FP0's f1 input
  copy_kernel<<<gdiv((long long)BATCH * NP0 * 3, 256), 256, 0, stream>>>(
      out1, xyz0, (long long)BATCH * NP0 * 3);
  cvt_f16_kernel<<<gdiv((long long)BATCH * NP0 * 3, 256), 256, 0, stream>>>(
      xyz0, xyz0h, (long long)BATCH * NP0 * 3);

  // ---- SA1: 4096 -> 1024, k=32, MLP 3->32->32->32 ----
  const long long R1 = (long long)BATCH * NP1 * NS1;
  fps_kernel<<<BATCH, 256, 0, stream>>>(xyz0, NP0, NP1, fidx1);
  gather3_kernel<<<gdiv((long long)BATCH * NP1, 256), 256, 0, stream>>>(xyz0, fidx1, NP0, NP1, xyz1);
  knn_kernel<<<dim3(NP1 / 64, BATCH), 64, 0, stream>>>(xyz0, NP0, xyz1, NP1, NS1, nidx1);
  group_rel_kernel<<<gdiv(R1, 256), 256, 0, stream>>>(xyz0, xyz1, nidx1, NP1, NS1, NP0, hA);
  run_mlp_layer(hA, 32, saP[0][0], sa[0][0], hB, R1, 32);
  run_mlp_layer(hB, 32, saP[0][1], sa[0][1], hA, R1, 32);
  run_mlp_layer(hA, 32, saP[0][2], sa[0][2], hB, R1, 32);
  maxpool_kernel<<<gdiv((long long)BATCH * NP1 * 32, 256), 256, 0, stream>>>(
      hB, (long long)BATCH * NP1, NS1, 32, feat1);

  // ---- SA2: 1024 -> 512, k=48, MLP 64->64->64->64 ----
  const long long R2 = (long long)BATCH * NP2 * NS2;
  fps_kernel<<<BATCH, 256, 0, stream>>>(xyz1, NP1, NP2, fidx2);
  gather3_kernel<<<gdiv((long long)BATCH * NP2, 256), 256, 0, stream>>>(xyz1, fidx2, NP1, NP2, xyz2);
  knn_kernel<<<dim3(NP2 / 64, BATCH), 64, 0, stream>>>(xyz1, NP1, xyz2, NP2, NS2, nidx2);
  group_feat_kernel<<<gdiv(R2, 256), 256, 0, stream>>>(feat1, nidx2, fidx2, NP2, NS2, NP1, 32, hA);
  run_mlp_layer(hA, 64, saP[1][0], sa[1][0], hB, R2, 64);
  run_mlp_layer(hB, 64, saP[1][1], sa[1][1], hA, R2, 64);
  run_mlp_layer(hA, 64, saP[1][2], sa[1][2], hB, R2, 64);
  maxpool_kernel<<<gdiv((long long)BATCH * NP2 * 64, 256), 256, 0, stream>>>(
      hB, (long long)BATCH * NP2, NS2, 64, feat2);

  // ---- SA3: 512 -> 256, k=48, MLP 128->128->128->128 ----
  const long long R3 = (long long)BATCH * NP3 * NS3;
  fps_kernel<<<BATCH, 256, 0, stream>>>(xyz2, NP2, NP3, fidx3);
  gather3_kernel<<<gdiv((long long)BATCH * NP3, 256), 256, 0, stream>>>(xyz2, fidx3, NP2, NP3, xyz3);
  knn_kernel<<<dim3(NP3 / 64, BATCH), 64, 0, stream>>>(xyz2, NP2, xyz3, NP3, NS3, nidx3);
  group_feat_kernel<<<gdiv(R3, 256), 256, 0, stream>>>(feat2, nidx3, fidx3, NP3, NS3, NP2, 64, hA);
  run_mlp_layer(hA, 128, saP[2][0], sa[2][0], hB, R3, 128);
  run_mlp_layer(hB, 128, saP[2][1], sa[2][1], hA, R3, 128);
  run_mlp_layer(hA, 128, saP[2][2], sa[2][2], hB, R3, 128);
  maxpool_kernel<<<gdiv((long long)BATCH * NP3 * 128, 256), 256, 0, stream>>>(
      hB, (long long)BATCH * NP3, NS3, 128, feat3);

  // ---- FP i=2: interp 256->512, cat 64+128=192, MLP 192->128->128 ----
  knn_kernel<<<dim3(NP2 / 64, BATCH), 64, 0, stream>>>(xyz3, NP3, xyz2, NP2, 3, fpidx2);
  fp_cat_kernel<<<gdiv((long long)BATCH * NP2, 256), 256, 0, stream>>>(
      xyz2, xyz3, fpidx2, feat2, 64, feat3, 128, NP2, NP3, 192, hA);
  run_mlp_layer(hA, 192, fpP[2][0], fp[2][0], hB, (long long)BATCH * NP2, 128);
  run_mlp_layer(hB, 128, fpP[2][1], fp[2][1], fpf2, (long long)BATCH * NP2, 128);

  // ---- FP i=1: interp 512->1024, cat 32+128=160, MLP 160->128->64 ----
  knn_kernel<<<dim3(NP1 / 64, BATCH), 64, 0, stream>>>(xyz2, NP2, xyz1, NP1, 3, fpidx1);
  fp_cat_kernel<<<gdiv((long long)BATCH * NP1, 256), 256, 0, stream>>>(
      xyz1, xyz2, fpidx1, feat1, 32, fpf2, 128, NP1, NP2, 160, hA);
  run_mlp_layer(hA, 160, fpP[1][0], fp[1][0], hB, (long long)BATCH * NP1, 128);
  run_mlp_layer(hB, 128, fpP[1][1], fp[1][1], fpf1, (long long)BATCH * NP1, 64);

  // ---- FP i=0: interp 1024->4096, cat 3+64=67 (pad 96), MLP 67->32->32 ----
  knn_kernel<<<dim3(NP0 / 64, BATCH), 64, 0, stream>>>(xyz1, NP1, xyz0, NP0, 3, fpidx0);
  fp_cat_kernel<<<gdiv((long long)BATCH * NP0, 256), 256, 0, stream>>>(
      xyz0, xyz1, fpidx0, xyz0h, 3, fpf1, 64, NP0, NP1, 96, hA);
  run_mlp_layer(hA, 96, fpP[0][0], fp[0][0], hB, (long long)BATCH * NP0, 32);
  run_mlp_layer(hB, 32, fpP[0][1], fp[0][1], fpf0, (long long)BATCH * NP0, 32);

  // ---- head ----
  // out2 = fpf0 @ cov_final.W + b (plain linear, f32 out)
  run_linear(fpf0, 32, covP, cov_b, out2buf, (long long)BATCH * NP0, 32, MMF_BIAS);
  // 10-NN covariance eigen features
  knn_kernel<<<dim3(NP0 / 64, BATCH), 64, 0, stream>>>(xyz0, NP0, xyz0, NP0, EDK, gidx);
  eig_kernel<<<gdiv((long long)BATCH * NP0, 256), 256, 0, stream>>>(
      xyz0, gidx, ed1_W, ed1_b, ed2_W, ed2_b, efbuf);
  concat_pad_kernel<<<gdiv((long long)BATCH * NP0, 256), 256, 0, stream>>>(out2buf, efbuf, zbuf);
  // conv1 (40->32, K padded 64) + bias + global BN(bn1) + relu + transpose
  zero_kernel<<<1, 256, 0, stream>>>(stats, 64);
  run_linear(zbuf, 64, conv1P, conv1_b, z2buf, (long long)BATCH * NP0, 32,
             MMF_BIAS | MMF_STATS);
  bn_finalize_kernel<<<1, 128, 0, stream>>>(stats, bn1_g, bn1_b, 32,
                                            1.0f / (float)((long long)BATCH * NP0));
  bn_relu_transpose_kernel<<<gdiv((long long)BATCH * NP0 * 32, 256), 256, 0, stream>>>(
      z2buf, stats, outz);
}
